// HybridKernelClassifier_65481071407685
// MI455X (gfx1250) — compile-verified, matrix operand order checked by reading, not run on hardware
//
#include <hip/hip_runtime.h>
#include <hip/hip_bf16.h>

// ---------------------------------------------------------------------------
// Hybrid CNN + quantum-fidelity-kernel classifier for MI455X (gfx1250).
// FC1 (64 x 55815 x 120) runs on the WMMA pipe in bf16 with f32 accumulate,
// 2x2 register tiling (32x32 output per wave), software-pipelined loads,
// split-K with non-atomic partial tiles + fused reduce/bias/ReLU.
// ---------------------------------------------------------------------------

typedef __attribute__((ext_vector_type(16))) __bf16 v16bf;
typedef __attribute__((ext_vector_type(8)))  __bf16 v8bf;
typedef __attribute__((ext_vector_type(8)))  float  v8f;

#define BATCH      64
#define C1_HW      124          // conv1 output spatial
#define P1_HW      123          // after 2x2 s1 pool
#define C2_HW      62           // conv2 output spatial
#define P2_HW      61           // after pool
#define FLAT_K     55815        // 15*61*61
#define K_PAD      55840        // round up to multiple of 32
#define K_ITERS    1745         // K_PAD / 32
#define KSPLIT     64
#define ITERS_PER  ((K_ITERS + KSPLIT - 1) / KSPLIT)   // 28
#define N_PAD      128          // 120 rounded to 8 tiles of 16

// ---------------------------------------------------------------------------
// conv1: 3->6, k5, s2, p1, +bias, ReLU.  out [64,6,124,124]
// ---------------------------------------------------------------------------
__global__ __launch_bounds__(256)
void conv1_kernel(const float* __restrict__ x, const float* __restrict__ w,
                  const float* __restrict__ bias, float* __restrict__ out) {
  int idx = blockIdx.x * blockDim.x + threadIdx.x;
  const int total = BATCH * 6 * C1_HW * C1_HW;
  if (idx >= total) return;
  int ox = idx % C1_HW;
  int oy = (idx / C1_HW) % C1_HW;
  int oc = (idx / (C1_HW * C1_HW)) % 6;
  int b  = idx / (C1_HW * C1_HW * 6);
  int iy0 = oy * 2 - 1, ix0 = ox * 2 - 1;
  float s = bias[oc];
  const float* wp = w + oc * 3 * 25;
  for (int ic = 0; ic < 3; ++ic) {
    const float* xp = x + (size_t)(b * 3 + ic) * (250 * 250);
#pragma unroll
    for (int ky = 0; ky < 5; ++ky) {
      int iy = iy0 + ky;
      if ((unsigned)iy < 250u) {
#pragma unroll
        for (int kx = 0; kx < 5; ++kx) {
          int ix = ix0 + kx;
          if ((unsigned)ix < 250u)
            s = fmaf(xp[iy * 250 + ix], wp[ic * 25 + ky * 5 + kx], s);
        }
      }
    }
  }
  out[idx] = s > 0.f ? s : 0.f;
}

// ---------------------------------------------------------------------------
// pool1: 2x2 stride1 max.  [64,6,124,124] -> [64,6,123,123]
// ---------------------------------------------------------------------------
__global__ __launch_bounds__(256)
void pool1_kernel(const float* __restrict__ in, float* __restrict__ out) {
  int idx = blockIdx.x * blockDim.x + threadIdx.x;
  const int total = BATCH * 6 * P1_HW * P1_HW;
  if (idx >= total) return;
  int ox = idx % P1_HW;
  int oy = (idx / P1_HW) % P1_HW;
  int c  = (idx / (P1_HW * P1_HW)) % 6;
  int b  = idx / (P1_HW * P1_HW * 6);
  const float* p = in + (size_t)(b * 6 + c) * (C1_HW * C1_HW) + oy * C1_HW + ox;
  out[idx] = fmaxf(fmaxf(p[0], p[1]), fmaxf(p[C1_HW], p[C1_HW + 1]));
}

// ---------------------------------------------------------------------------
// conv2: 6->15, k3, s2, p1, +bias, ReLU.  [64,6,123,123] -> [64,15,62,62]
// ---------------------------------------------------------------------------
__global__ __launch_bounds__(256)
void conv2_kernel(const float* __restrict__ in, const float* __restrict__ w,
                  const float* __restrict__ bias, float* __restrict__ out) {
  int idx = blockIdx.x * blockDim.x + threadIdx.x;
  const int total = BATCH * 15 * C2_HW * C2_HW;
  if (idx >= total) return;
  int ox = idx % C2_HW;
  int oy = (idx / C2_HW) % C2_HW;
  int oc = (idx / (C2_HW * C2_HW)) % 15;
  int b  = idx / (C2_HW * C2_HW * 15);
  int iy0 = oy * 2 - 1, ix0 = ox * 2 - 1;
  float s = bias[oc];
  const float* wp = w + oc * 6 * 9;
  for (int ic = 0; ic < 6; ++ic) {
    const float* xp = in + (size_t)(b * 6 + ic) * (P1_HW * P1_HW);
#pragma unroll
    for (int ky = 0; ky < 3; ++ky) {
      int iy = iy0 + ky;
      if ((unsigned)iy < (unsigned)P1_HW) {
#pragma unroll
        for (int kx = 0; kx < 3; ++kx) {
          int ix = ix0 + kx;
          if ((unsigned)ix < (unsigned)P1_HW)
            s = fmaf(xp[iy * P1_HW + ix], wp[ic * 9 + ky * 3 + kx], s);
        }
      }
    }
  }
  out[idx] = s > 0.f ? s : 0.f;
}

// ---------------------------------------------------------------------------
// pool2 + flatten + bf16 convert.  [64,15,62,62] -> act[64, K_PAD] (bf16)
// ---------------------------------------------------------------------------
__global__ __launch_bounds__(256)
void pool2_cvt_kernel(const float* __restrict__ in, __hip_bfloat16* __restrict__ act) {
  int idx = blockIdx.x * blockDim.x + threadIdx.x;
  const int total = BATCH * FLAT_K;
  if (idx >= total) return;
  int j = idx % FLAT_K;
  int b = idx / FLAT_K;
  int c = j / (P2_HW * P2_HW);
  int r = j % (P2_HW * P2_HW);
  int y = r / P2_HW, xq = r % P2_HW;
  const float* p = in + (size_t)(b * 15 + c) * (C2_HW * C2_HW) + y * C2_HW + xq;
  float m = fmaxf(fmaxf(p[0], p[1]), fmaxf(p[C2_HW], p[C2_HW + 1]));
  act[(size_t)b * K_PAD + j] = __float2bfloat16(m);
}

__global__ __launch_bounds__(256)
void pad_act_kernel(__hip_bfloat16* __restrict__ act) {
  int idx = blockIdx.x * blockDim.x + threadIdx.x;
  const int padw = K_PAD - FLAT_K;             // 25
  if (idx >= BATCH * padw) return;
  int b = idx / padw, j = idx % padw;
  act[(size_t)b * K_PAD + FLAT_K + j] = __float2bfloat16(0.f);
}

// ---------------------------------------------------------------------------
// fc1 weights -> bf16, padded to [N_PAD, K_PAD] with zeros
// ---------------------------------------------------------------------------
__global__ __launch_bounds__(256)
void cvt_w1_kernel(const float* __restrict__ w, __hip_bfloat16* __restrict__ wb) {
  size_t idx = (size_t)blockIdx.x * blockDim.x + threadIdx.x;
  const size_t total = (size_t)N_PAD * K_PAD;
  if (idx >= total) return;
  int o = (int)(idx / K_PAD);
  int k = (int)(idx % K_PAD);
  float v = (o < 120 && k < FLAT_K) ? w[(size_t)o * FLAT_K + k] : 0.f;
  wb[idx] = __float2bfloat16(v);
}

// ---------------------------------------------------------------------------
// FC1 GEMM on the WMMA pipe.
//   C[64,128] = A[64,K_PAD](bf16) * B[128,K_PAD]^T
// One wave per 32x32 output tile per K-split (2x2 register tile, 4 WMMAs per
// K-step). Loads for step i+1 are issued before the WMMAs of step i so the
// matrix pipe runs with 8 b128 loads in flight. Partial tiles go to a
// [KSPLIT,64,128] buffer (no atomics); a reduce kernel folds splits + bias.
// Fragment layout per ISA 7.12.2 (16-bit A 16x32): lanes 0-15 hold row
// M=lane with K {0..7,16..23}; lanes 16-31 hold K {8..15,24..31}; B mirrors
// for columns. C: lane&15 = N, vgpr r = M (+8 for upper half-wave).
// ---------------------------------------------------------------------------
__device__ __forceinline__ v8bf ld8(const __bf16* p) {
  return *reinterpret_cast<const v8bf*>(p);
}

__global__ __launch_bounds__(32)
void fc1_wmma_kernel(const __hip_bfloat16* __restrict__ act_h,
                     const __hip_bfloat16* __restrict__ w1_h,
                     float* __restrict__ part) {
  const __bf16* __restrict__ A = reinterpret_cast<const __bf16*>(act_h);
  const __bf16* __restrict__ Bm = reinterpret_cast<const __bf16*>(w1_h);

  const int tm2 = blockIdx.x >> 2;    // 0..1 : M rows [tm2*32, +32)
  const int tn2 = blockIdx.x & 3;     // 0..3 : N cols [tn2*32, +32)
  const int lane = threadIdx.x;       // 0..31
  const int half8 = (lane >> 4) * 8;  // 0 or 8
  const int l15 = lane & 15;

  const __bf16* a0row = A  + (size_t)(tm2 * 32 + l15) * K_PAD;
  const __bf16* a1row = a0row + (size_t)16 * K_PAD;
  const __bf16* b0row = Bm + (size_t)(tn2 * 32 + l15) * K_PAD;
  const __bf16* b1row = b0row + (size_t)16 * K_PAD;

  int it0 = blockIdx.y * ITERS_PER;
  int it1 = it0 + ITERS_PER;
  if (it1 > K_ITERS) it1 = K_ITERS;

  v8f acc00 = {0.f,0.f,0.f,0.f,0.f,0.f,0.f,0.f};
  v8f acc01 = acc00, acc10 = acc00, acc11 = acc00;

  // preload first fragments (clamped so empty tail splits read valid memory)
  int kp = ((it0 < K_ITERS) ? it0 : (K_ITERS - 1)) * 32 + half8;
  v8bf a0l = ld8(a0row + kp), a0h = ld8(a0row + kp + 16);
  v8bf a1l = ld8(a1row + kp), a1h = ld8(a1row + kp + 16);
  v8bf b0l = ld8(b0row + kp), b0h = ld8(b0row + kp + 16);
  v8bf b1l = ld8(b1row + kp), b1h = ld8(b1row + kp + 16);

  for (int it = it0; it < it1; ++it) {
    const int itn = (it + 1 < it1) ? it + 1 : it;
    const int kn = itn * 32 + half8;
    // stream ~16 K-steps ahead on all four operand rows
    __builtin_prefetch(a0row + kn + 512, 0, 1);
    __builtin_prefetch(a1row + kn + 512, 0, 1);
    __builtin_prefetch(b0row + kn + 512, 0, 1);
    __builtin_prefetch(b1row + kn + 512, 0, 1);
    // issue next-step loads before consuming the current fragments
    v8bf na0l = ld8(a0row + kn), na0h = ld8(a0row + kn + 16);
    v8bf na1l = ld8(a1row + kn), na1h = ld8(a1row + kn + 16);
    v8bf nb0l = ld8(b0row + kn), nb0h = ld8(b0row + kn + 16);
    v8bf nb1l = ld8(b1row + kn), nb1h = ld8(b1row + kn + 16);

    v16bf A0, A1, B0, B1;
#pragma unroll
    for (int i = 0; i < 8; ++i) {
      A0[i] = a0l[i]; A0[i + 8] = a0h[i];
      A1[i] = a1l[i]; A1[i + 8] = a1h[i];
      B0[i] = b0l[i]; B0[i + 8] = b0h[i];
      B1[i] = b1l[i]; B1[i + 8] = b1h[i];
    }
    acc00 = __builtin_amdgcn_wmma_f32_16x16x32_bf16(false, A0, false, B0, (short)0, acc00, false, false);
    acc01 = __builtin_amdgcn_wmma_f32_16x16x32_bf16(false, A0, false, B1, (short)0, acc01, false, false);
    acc10 = __builtin_amdgcn_wmma_f32_16x16x32_bf16(false, A1, false, B0, (short)0, acc10, false, false);
    acc11 = __builtin_amdgcn_wmma_f32_16x16x32_bf16(false, A1, false, B1, (short)0, acc11, false, false);

    a0l = na0l; a0h = na0h; a1l = na1l; a1h = na1h;
    b0l = nb0l; b0h = nb0h; b1l = nb1l; b1h = nb1h;
  }

  // store the 32x32 partial tile (empty splits store zeros -> reduce is exact)
  float* base = part + (size_t)blockIdx.y * (BATCH * N_PAD);
  const int m0 = tm2 * 32 + half8;
  const int n0 = tn2 * 32 + l15;
#pragma unroll
  for (int r = 0; r < 8; ++r) {
    base[(m0 + r) * N_PAD + n0]            = acc00[r];
    base[(m0 + r) * N_PAD + n0 + 16]       = acc01[r];
    base[(m0 + 16 + r) * N_PAD + n0]       = acc10[r];
    base[(m0 + 16 + r) * N_PAD + n0 + 16]  = acc11[r];
  }
}

// ---------------------------------------------------------------------------
// split-K reduce + bias + ReLU:  part[KSPLIT,64,128] -> fc1out[64,120]
// ---------------------------------------------------------------------------
__global__ __launch_bounds__(256)
void fc1_reduce_kernel(const float* __restrict__ part, const float* __restrict__ bias,
                       float* __restrict__ out) {
  int idx = blockIdx.x * blockDim.x + threadIdx.x;
  if (idx >= BATCH * N_PAD) return;
  float s = 0.f;
  for (int sp = 0; sp < KSPLIT; ++sp)
    s += part[(size_t)sp * (BATCH * N_PAD) + idx];
  int n = idx & (N_PAD - 1);
  int b = idx >> 7;
  if (n < 120) {
    float v = s + bias[n];
    out[b * 120 + n] = v > 0.f ? v : 0.f;
  }
}

// ---------------------------------------------------------------------------
// FC2 (120->84, ReLU) + quantum fidelity kernel over first 4 features.
// One block per batch element. kmat[64,64]
// ---------------------------------------------------------------------------
__global__ __launch_bounds__(128)
void fc2_qkernel(const float* __restrict__ fc1out, const float* __restrict__ w2,
                 const float* __restrict__ b2, const float* __restrict__ centers,
                 float* __restrict__ kmat) {
  __shared__ float feat[84];
  const int b = blockIdx.x, t = threadIdx.x;
  if (t < 84) {
    float s = b2[t];
    const float* h  = fc1out + b * 120;
    const float* wr = w2 + t * 120;
    for (int j = 0; j < 120; ++j) s = fmaf(h[j], wr[j], s);
    feat[t] = s > 0.f ? s : 0.f;
  }
  __syncthreads();
  if (t < 64) {
    const float* c = centers + t * 84;
    float p = 1.f;
#pragma unroll
    for (int i = 0; i < 4; ++i) p *= cosf(0.5f * (feat[i] - c[i]));
    kmat[b * 64 + t] = fabsf(p);
  }
}

// ---------------------------------------------------------------------------
// Classifier head: two 64x64 ReLU layers, 64->2 head, softmax. One block.
// ---------------------------------------------------------------------------
__global__ __launch_bounds__(256)
void cls_kernel(const float* __restrict__ kmat,
                const float* __restrict__ w1, const float* __restrict__ b1,
                const float* __restrict__ w2, const float* __restrict__ b2,
                const float* __restrict__ w3, const float* __restrict__ b3,
                float* __restrict__ out) {
  __shared__ float sk[64 * 64];
  __shared__ float s1[64 * 64];
  __shared__ float s2[64 * 64];
  __shared__ float slog[128];
  const int t = threadIdx.x;
  for (int i = t; i < 64 * 64; i += 256) sk[i] = kmat[i];
  __syncthreads();
  for (int idx = t; idx < 64 * 64; idx += 256) {
    int b = idx >> 6, o = idx & 63;
    float s = b1[o];
    const float* wr = w1 + o * 64;
    for (int j = 0; j < 64; ++j) s = fmaf(sk[b * 64 + j], wr[j], s);
    s1[idx] = s > 0.f ? s : 0.f;
  }
  __syncthreads();
  for (int idx = t; idx < 64 * 64; idx += 256) {
    int b = idx >> 6, o = idx & 63;
    float s = b2[o];
    const float* wr = w2 + o * 64;
    for (int j = 0; j < 64; ++j) s = fmaf(s1[b * 64 + j], wr[j], s);
    s2[idx] = s > 0.f ? s : 0.f;
  }
  __syncthreads();
  if (t < 128) {
    int b = t >> 1, o = t & 1;
    float s = b3[o];
    const float* wr = w3 + o * 64;
    for (int j = 0; j < 64; ++j) s = fmaf(s2[b * 64 + j], wr[j], s);
    slog[t] = s;
  }
  __syncthreads();
  if (t < 64) {
    float l0 = slog[t * 2], l1 = slog[t * 2 + 1];
    float m = fmaxf(l0, l1);
    float e0 = expf(l0 - m), e1 = expf(l1 - m);
    float inv = 1.f / (e0 + e1);
    out[t * 2]     = e0 * inv;
    out[t * 2 + 1] = e1 * inv;
  }
}

// ---------------------------------------------------------------------------
// Launch
// ---------------------------------------------------------------------------
extern "C" void kernel_launch(void* const* d_in, const int* in_sizes, int n_in,
                              void* d_out, int out_size, void* d_ws, size_t ws_size,
                              hipStream_t stream) {
  const float* x   = (const float*)d_in[0];
  const float* c1w = (const float*)d_in[1];
  const float* c1b = (const float*)d_in[2];
  const float* c2w = (const float*)d_in[3];
  const float* c2b = (const float*)d_in[4];
  const float* f1w = (const float*)d_in[5];
  const float* f1b = (const float*)d_in[6];
  const float* f2w = (const float*)d_in[7];
  const float* f2b = (const float*)d_in[8];
  const float* cen = (const float*)d_in[9];
  const float* cw1 = (const float*)d_in[10];
  const float* cb1 = (const float*)d_in[11];
  const float* cw2 = (const float*)d_in[12];
  const float* cb2 = (const float*)d_in[13];
  const float* cw3 = (const float*)d_in[14];
  const float* cb3 = (const float*)d_in[15];

  char* ws = (char*)d_ws;
  size_t off = 0;
  auto take = [&](size_t bytes) -> char* {
    char* p = ws + off;
    off = (off + bytes + 255) & ~(size_t)255;
    return p;
  };
  float*          c1out  = (float*)take((size_t)BATCH * 6 * C1_HW * C1_HW * 4);
  float*          p1out  = (float*)take((size_t)BATCH * 6 * P1_HW * P1_HW * 4);
  float*          c2out  = (float*)take((size_t)BATCH * 15 * C2_HW * C2_HW * 4);
  __hip_bfloat16* act    = (__hip_bfloat16*)take((size_t)BATCH * K_PAD * 2);
  __hip_bfloat16* w1b    = (__hip_bfloat16*)take((size_t)N_PAD * K_PAD * 2);
  float*          part   = (float*)take((size_t)KSPLIT * BATCH * N_PAD * 4);
  float*          fc1out = (float*)take((size_t)BATCH * 120 * 4);
  float*          kmat   = (float*)take((size_t)BATCH * 64 * 4);
  (void)ws_size; (void)in_sizes; (void)n_in; (void)out_size;

  auto blocks = [](long long n) { return (unsigned)((n + 255) / 256); };

  // independent weight conversion first (cheap, fills L2 with bf16 weights)
  cvt_w1_kernel<<<blocks((long long)N_PAD * K_PAD), 256, 0, stream>>>(f1w, w1b);

  conv1_kernel<<<blocks((long long)BATCH * 6 * C1_HW * C1_HW), 256, 0, stream>>>(x, c1w, c1b, c1out);
  pool1_kernel<<<blocks((long long)BATCH * 6 * P1_HW * P1_HW), 256, 0, stream>>>(c1out, p1out);
  conv2_kernel<<<blocks((long long)BATCH * 15 * C2_HW * C2_HW), 256, 0, stream>>>(p1out, c2w, c2b, c2out);
  pool2_cvt_kernel<<<blocks((long long)BATCH * FLAT_K), 256, 0, stream>>>(c2out, act);
  pad_act_kernel<<<blocks((long long)BATCH * (K_PAD - FLAT_K)), 256, 0, stream>>>(act);

  dim3 gemm_grid(8, KSPLIT);           // 8 output tiles (2x4) x 64 K-splits
  fc1_wmma_kernel<<<gemm_grid, 32, 0, stream>>>(act, w1b, part);

  fc1_reduce_kernel<<<blocks(BATCH * N_PAD), 256, 0, stream>>>(part, f1b, fc1out);
  fc2_qkernel<<<BATCH, 128, 0, stream>>>(fc1out, f2w, f2b, cen, kmat);
  cls_kernel<<<1, 256, 0, stream>>>(kmat, cw1, cb1, cw2, cb2, cw3, cb3, (float*)d_out);
}
